// SimplifiedRetNet_54546084659884
// MI455X (gfx1250) — matrix operand order repaired and checked by reading, not compiled, not run on hardware
//
#include <hip/hip_runtime.h>
#include <cmath>

typedef __attribute__((ext_vector_type(16))) _Float16 v16h;
typedef __attribute__((ext_vector_type(8)))  _Float16 v8h;
typedef __attribute__((ext_vector_type(4)))  _Float16 v4h;
typedef __attribute__((ext_vector_type(8)))  float    v8f;
typedef __attribute__((ext_vector_type(4)))  float    v4f;
typedef __attribute__((ext_vector_type(4)))  int      v4i;

#define EP_F16    0
#define EP_DECAY  1
#define EP_GELU   2
#define EP_RESF32 3

#define BB 4
#define SS 2048
#define DD 2048
#define HH 2048
#define LN_EPS 1e-5f

// ----------------------------------------------------- async global -> LDS
// CDNA5 GLOBAL_LOAD_ASYNC_TO_LDS_B128: per-lane 16B DMA into LDS, ASYNCcnt.
__device__ __forceinline__ void async_copy_b128(const _Float16* g, _Float16* l) {
#if __has_builtin(__builtin_amdgcn_global_load_async_to_lds_b128)
  typedef __attribute__((address_space(1))) v4i* gp_t;
  typedef __attribute__((address_space(3))) v4i* lp_t;
  __builtin_amdgcn_global_load_async_to_lds_b128(
      (gp_t)(void*)g, (lp_t)(void*)l, 0, 0);
#else
  unsigned loff = (unsigned)(unsigned long long)l;   // LDS aperture: low 32 bits
  asm volatile("global_load_async_to_lds_b128 %0, %1, off"
               :: "v"(loff), "v"(g) : "memory");
#endif
}

__device__ __forceinline__ void wait_async0() {
#if __has_builtin(__builtin_amdgcn_s_wait_asynccnt)
  __builtin_amdgcn_s_wait_asynccnt(0);
#else
  asm volatile("s_wait_asynccnt 0x0" ::: "memory");
#endif
}

// ---------------------------------------------------------------- f32 -> f16
__global__ __launch_bounds__(256)
void cvt_f32_f16_kernel(const float* __restrict__ in, _Float16* __restrict__ out, long n) {
  long i = ((long)blockIdx.x * 256 + threadIdx.x) * 4;
  if (i >= n) return;
  v4f v = *(const v4f*)(in + i);
  v4h o;
  #pragma unroll
  for (int e = 0; e < 4; ++e) o[e] = (_Float16)v[e];
  *(v4h*)(out + i) = o;
}

// ------------------------------------------------------- f16 64x64 transpose
__global__ __launch_bounds__(256)
void transpose_f16_kernel(const _Float16* __restrict__ in, _Float16* __restrict__ out,
                          int S, int H) {
  __shared__ _Float16 tile[64][65];
  const long boff = (long)blockIdx.z * (long)S * (long)H;
  const int hs = blockIdx.x * 64;
  const int ss = blockIdx.y * 64;
  #pragma unroll
  for (int i = 0; i < 16; ++i) {
    int idx = threadIdx.x + 256 * i;
    int r = idx >> 6, c = idx & 63;
    tile[r][c] = in[boff + (long)(ss + r) * H + hs + c];
  }
  __syncthreads();
  #pragma unroll
  for (int i = 0; i < 16; ++i) {
    int idx = threadIdx.x + 256 * i;
    int r = idx >> 6, c = idx & 63;                 // r: h offset, c: s offset
    out[boff + (long)(hs + r) * S + ss + c] = tile[c][r];
  }
}

// --------------------------------------------------------------- WMMA GEMM
// C[M,N] = A[M,K] * Bt[N,K]^T   (row-major f16 operands, f32 accumulate)
// Block tile 128x128x32, 8 waves (4 along M x 2 along N), wave tile 32x64.
// Double-buffered LDS filled by GLOBAL_LOAD_ASYNC_TO_LDS_B128.
template<int EP>
__global__ __launch_bounds__(256)
void gemm_wmma_kernel(const _Float16* __restrict__ A, long strideA,
                      const _Float16* __restrict__ Bt, long strideB,
                      void* __restrict__ Cv, long strideC,
                      int M, int N, int K,
                      const float* __restrict__ bias,
                      const float* __restrict__ decay,
                      const float* __restrict__ resid,
                      float scale, int triangular) {
  __shared__ __align__(16) _Float16 sA[2][128 * 40];  // 32 halfs + 8 pad per row
  __shared__ __align__(16) _Float16 sB[2][128 * 40];

  const int t    = threadIdx.x;
  const int lane = t & 31;
  const int w    = t >> 5;
  const int wm   = w & 3;           // wave position along M (0..3)
  const int wn   = w >> 2;          // wave position along N (0..1)
  const int g    = lane >> 4;       // lane half-group
  const int ln   = lane & 15;

  const int bM = blockIdx.y * 128;
  const int bN = blockIdx.x * 128;
  const int bz = blockIdx.z;

  A  += (long)bz * strideA;
  Bt += (long)bz * strideB;

  int kIter = K >> 5;
  if (EP == EP_DECAY && bN > bM + 127) kIter = 0;          // fully above diagonal
  if (triangular) {                                        // scores@V: k <= q only
    int kmax = bM + 128; if (kmax > K) kmax = K;
    int ki = (kmax + 31) >> 5; if (ki < kIter) kIter = ki;
  }

  // per-thread staging slots: 512 chunks of 8 halfs per 128x32 tile
  const int r0  = t >> 2;
  const int cc0 = (t & 3) << 3;
  const int r1  = (t + 256) >> 2;
  const int cc1 = ((t + 256) & 3) << 3;

  v8f acc[2][4];
  #pragma unroll
  for (int i = 0; i < 2; ++i)
    #pragma unroll
    for (int j = 0; j < 4; ++j)
      #pragma unroll
      for (int e = 0; e < 8; ++e) acc[i][j][e] = 0.0f;

  if (kIter > 0) {                                   // prologue: tile 0 -> buf 0
    async_copy_b128(A  + (long)(bM + r0) * K + cc0, &sA[0][r0 * 40 + cc0]);
    async_copy_b128(Bt + (long)(bN + r0) * K + cc0, &sB[0][r0 * 40 + cc0]);
    async_copy_b128(A  + (long)(bM + r1) * K + cc1, &sA[0][r1 * 40 + cc1]);
    async_copy_b128(Bt + (long)(bN + r1) * K + cc1, &sB[0][r1 * 40 + cc1]);
  }

  for (int kt = 0; kt < kIter; ++kt) {
    wait_async0();          // this wave's tile-kt pieces are in LDS
    __syncthreads();        // everyone's are; buf 1-p free (read last iter, pre-barrier)

    if (kt + 1 < kIter) {   // async-stage tile kt+1 into the other buffer
      const int nb = (kt + 1) & 1;
      const int k1 = (kt + 1) << 5;
      async_copy_b128(A  + (long)(bM + r0) * K + k1 + cc0, &sA[nb][r0 * 40 + cc0]);
      async_copy_b128(Bt + (long)(bN + r0) * K + k1 + cc0, &sB[nb][r0 * 40 + cc0]);
      async_copy_b128(A  + (long)(bM + r1) * K + k1 + cc1, &sA[nb][r1 * 40 + cc1]);
      async_copy_b128(Bt + (long)(bN + r1) * K + k1 + cc1, &sB[nb][r1 * 40 + cc1]);
    }
    if (kt + 2 < kIter) {   // L2 prefetch two tiles ahead (global_prefetch_b8)
      const int k2 = (kt + 2) << 5;
      __builtin_prefetch(A  + (long)(bM + r0) * K + k2 + cc0, 0, 1);
      __builtin_prefetch(Bt + (long)(bN + r0) * K + k2 + cc0, 0, 1);
    }

    const int p = kt & 1;
    // -------- fragments per ISA 7.12.2 wave32 layouts
    v16h af[2], bf[4];
    #pragma unroll
    for (int fm = 0; fm < 2; ++fm) {
      const _Float16* pa = &sA[p][(wm * 32 + fm * 16 + ln) * 40 + 8 * g];
      v8h lo = *(const v8h*)pa;
      v8h hi = *(const v8h*)(pa + 16);
      #pragma unroll
      for (int e = 0; e < 8; ++e) { af[fm][e] = lo[e]; af[fm][8 + e] = hi[e]; }
    }
    #pragma unroll
    for (int fn = 0; fn < 4; ++fn) {
      const _Float16* pb = &sB[p][(wn * 64 + fn * 16 + ln) * 40 + 8 * g];
      v8h lo = *(const v8h*)pb;
      v8h hi = *(const v8h*)(pb + 16);
      #pragma unroll
      for (int e = 0; e < 8; ++e) { bf[fn][e] = lo[e]; bf[fn][8 + e] = hi[e]; }
    }
    // -------- 8 WMMAs per wave per K-step
    #pragma unroll
    for (int fm = 0; fm < 2; ++fm)
      #pragma unroll
      for (int fn = 0; fn < 4; ++fn)
        acc[fm][fn] = __builtin_amdgcn_wmma_f32_16x16x32_f16(
            false, af[fm], false, bf[fn], (short)0, acc[fm][fn], false, false);
  }

  // -------- epilogue: C/D layout: VGPR r -> row r + 8*g, lane -> col
  #pragma unroll
  for (int fm = 0; fm < 2; ++fm) {
    #pragma unroll
    for (int fn = 0; fn < 4; ++fn) {
      const int rbase = bM + wm * 32 + fm * 16 + 8 * g;
      const int col   = bN + wn * 64 + fn * 16 + ln;
      float dj = 0.0f, bj = 0.0f;
      if (EP == EP_DECAY) dj = decay[col];
      if (EP == EP_GELU || EP == EP_RESF32) bj = bias[col];
      #pragma unroll
      for (int r = 0; r < 8; ++r) {
        const int row = rbase + r;
        float v = acc[fm][fn][r];
        if (EP == EP_F16) {
          ((_Float16*)Cv)[(long)bz * strideC + (long)row * N + col] = (_Float16)v;
        } else if (EP == EP_DECAY) {
          float m = (col <= row) ? expf(-dj * (float)(row - col)) : 0.0f;
          v = v * m * scale;
          ((_Float16*)Cv)[(long)bz * strideC + (long)row * N + col] = (_Float16)v;
        } else if (EP == EP_GELU) {
          v += bj;
          v = 0.5f * v * (1.0f + erff(v * 0.70710678118654752f));
          ((_Float16*)Cv)[(long)row * N + col] = (_Float16)v;
        } else { // EP_RESF32: bias + residual, f32 out
          v += bj + resid[(long)row * N + col];
          ((float*)Cv)[(long)row * N + col] = v;
        }
      }
    }
  }
}

// --------------------------------------------------------------- LayerNorm
__global__ __launch_bounds__(256)
void layernorm_kernel(const float* __restrict__ Y, const float* __restrict__ gam,
                      const float* __restrict__ bet, float* __restrict__ out, int Dim) {
  __shared__ float red[256];
  const long row = blockIdx.x;
  const float* y = Y + row * (long)Dim;
  const int t = threadIdx.x;

  float s = 0.0f;
  for (int i = t; i < Dim; i += 256) s += y[i];
  red[t] = s; __syncthreads();
  for (int st = 128; st > 0; st >>= 1) { if (t < st) red[t] += red[t + st]; __syncthreads(); }
  const float mu = red[0] / (float)Dim;
  __syncthreads();

  float vs = 0.0f;
  for (int i = t; i < Dim; i += 256) { float d = y[i] - mu; vs += d * d; }
  red[t] = vs; __syncthreads();
  for (int st = 128; st > 0; st >>= 1) { if (t < st) red[t] += red[t + st]; __syncthreads(); }
  const float rstd = rsqrtf(red[0] / (float)Dim + LN_EPS);

  for (int i = t; i < Dim; i += 256)
    out[row * (long)Dim + i] = (y[i] - mu) * rstd * gam[i] + bet[i];
}

// ----------------------------------------------------------------- launcher
extern "C" void kernel_launch(void* const* d_in, const int* in_sizes, int n_in,
                              void* d_out, int out_size, void* d_ws, size_t ws_size,
                              hipStream_t stream) {
  const float* x    = (const float*)d_in[0];
  const float* Wq   = (const float*)d_in[1];
  const float* Wk   = (const float*)d_in[2];
  const float* Wv   = (const float*)d_in[3];
  const float* dec  = (const float*)d_in[4];
  const float* W1   = (const float*)d_in[5];
  const float* b1   = (const float*)d_in[6];
  const float* W2   = (const float*)d_in[7];
  const float* b2   = (const float*)d_in[8];
  const float* ln_g = (const float*)d_in[9];
  const float* ln_b = (const float*)d_in[10];

  char* ws = (char*)d_ws;
  const size_t MB = 1ull << 20;
  _Float16* xh   = (_Float16*)(ws + 0);        // 32 MB: x f16      (later: scores)
  _Float16* wbuf = (_Float16*)(ws + 32 * MB);  // 16 MB: weight f16 (reused)
  _Float16* Qh   = (_Float16*)(ws + 48 * MB);  // 32 MB            (later: attn out)
  _Float16* Kh   = (_Float16*)(ws + 80 * MB);  // 32 MB            (later: U low)
  _Float16* Vh   = (_Float16*)(ws + 112 * MB); // 32 MB            (later: U high)
  _Float16* Vth  = (_Float16*)(ws + 144 * MB); // 32 MB: V^T
  float*    Yf   = (float*)(ws + 176 * MB);    // 64 MB: pre-LN f32
  _Float16* scores = xh;
  _Float16* outh   = Qh;
  _Float16* U      = Kh;                        // 64 MB spans Kh+Vh

  const int  Mbs   = BB * SS;                   // 8192
  const long n_x   = (long)BB * SS * DD;
  const float qk_scale = 1.0f / sqrtf((float)HH);

  // x -> f16
  cvt_f32_f16_kernel<<<dim3((unsigned)(n_x / 1024)), dim3(256), 0, stream>>>(x, xh, n_x);

  // Q = x Wq^T ; K = x Wk^T ; V = x Wv^T
  const long n_w = (long)HH * DD;
  cvt_f32_f16_kernel<<<dim3((unsigned)(n_w / 1024)), dim3(256), 0, stream>>>(Wq, wbuf, n_w);
  gemm_wmma_kernel<EP_F16><<<dim3(HH / 128, Mbs / 128, 1), 256, 0, stream>>>(
      xh, 0, wbuf, 0, Qh, 0, Mbs, HH, DD, nullptr, nullptr, nullptr, 1.0f, 0);
  cvt_f32_f16_kernel<<<dim3((unsigned)(n_w / 1024)), dim3(256), 0, stream>>>(Wk, wbuf, n_w);
  gemm_wmma_kernel<EP_F16><<<dim3(HH / 128, Mbs / 128, 1), 256, 0, stream>>>(
      xh, 0, wbuf, 0, Kh, 0, Mbs, HH, DD, nullptr, nullptr, nullptr, 1.0f, 0);
  cvt_f32_f16_kernel<<<dim3((unsigned)(n_w / 1024)), dim3(256), 0, stream>>>(Wv, wbuf, n_w);
  gemm_wmma_kernel<EP_F16><<<dim3(HH / 128, Mbs / 128, 1), 256, 0, stream>>>(
      xh, 0, wbuf, 0, Vh, 0, Mbs, HH, DD, nullptr, nullptr, nullptr, 1.0f, 0);

  // V^T per batch
  transpose_f16_kernel<<<dim3(HH / 64, SS / 64, BB), 256, 0, stream>>>(Vh, Vth, SS, HH);

  // scores = (Q K^T) * exp(-decay[j]*(i-j)) * tril / sqrt(H)   (batched)
  gemm_wmma_kernel<EP_DECAY><<<dim3(SS / 128, SS / 128, BB), 256, 0, stream>>>(
      Qh, (long)SS * HH, Kh, (long)SS * HH, scores, (long)SS * SS,
      SS, SS, HH, nullptr, dec, nullptr, qk_scale, 0);

  // out = scores V   (batched, K-loop clipped at causal diagonal)
  gemm_wmma_kernel<EP_F16><<<dim3(HH / 128, SS / 128, BB), 256, 0, stream>>>(
      scores, (long)SS * SS, Vth, (long)HH * SS, outh, (long)SS * HH,
      SS, HH, SS, nullptr, nullptr, nullptr, 1.0f, 1);

  // FFN1: U = gelu(out W1^T + b1)
  const long n_w1 = 2L * DD * HH;
  cvt_f32_f16_kernel<<<dim3((unsigned)(n_w1 / 1024)), dim3(256), 0, stream>>>(W1, wbuf, n_w1);
  gemm_wmma_kernel<EP_GELU><<<dim3(2 * DD / 128, Mbs / 128, 1), 256, 0, stream>>>(
      outh, 0, wbuf, 0, U, 0, Mbs, 2 * DD, HH, b1, nullptr, nullptr, 1.0f, 0);

  // FFN2 + residual: Y = U W2^T + b2 + x   (f32)
  const long n_w2 = (long)DD * 2 * DD;
  cvt_f32_f16_kernel<<<dim3((unsigned)(n_w2 / 1024)), dim3(256), 0, stream>>>(W2, wbuf, n_w2);
  gemm_wmma_kernel<EP_RESF32><<<dim3(DD / 128, Mbs / 128, 1), 256, 0, stream>>>(
      U, 0, wbuf, 0, Yf, 0, Mbs, DD, 2 * DD, b2, nullptr, x, 1.0f, 0);

  // LayerNorm -> d_out
  layernorm_kernel<<<dim3(Mbs), dim3(256), 0, stream>>>(Yf, ln_g, ln_b, (float*)d_out, DD);
}
